// PConv_80066780332162
// MI455X (gfx1250) — matrix operand order repaired and checked by reading, not compiled, not run on hardware
//
#include <hip/hip_runtime.h>
#include <hip/hip_bf16.h>

typedef __attribute__((ext_vector_type(16))) __bf16  v16bf;
typedef __attribute__((ext_vector_type(8)))  __bf16  v8bf;
typedef __attribute__((ext_vector_type(8)))  float   v8f;
typedef __attribute__((ext_vector_type(4)))  unsigned u32x4;
typedef __attribute__((ext_vector_type(8)))  int      i32x8;
typedef __attribute__((ext_vector_type(4)))  int      i32x4;

#define PC_EPS 1e-5f

// Problem dims (fixed by the reference)
constexpr int Bn   = 4;
constexpr int Npts = 32768;
constexpr int Cch  = 64;      // feature channels
constexpr int NSm  = 16;      // neighbors per point
constexpr int Oout = 128;     // output channels
constexpr int Wch  = 64;      // weightnet channels
constexpr int PTS  = 16;      // points per workgroup
constexpr int KAGG = Cch * Wch;       // 4096

// ---- workspace layout (ushort/bf16 elements) ----
constexpr size_t WS_WF   = 0;                                  // 128*4096
constexpr size_t WS_W2   = WS_WF + (size_t)Oout * KAGG;        // 64*64
constexpr size_t WS_W3   = WS_W2 + 64 * 64;                    // 64*64
constexpr size_t WS_FEAT = WS_W3 + 64 * 64;                    // B*C*N

// ---- LDS layout (byte offsets into dynamic shared) ----
constexpr int LDS_FOLD = 0;        // 576 floats: inv1,sh1,inv2,sh2,b3,invF,shF
constexpr int LDS_IDX  = 2560;     // 256 * int
constexpr int LDS_GX   = 3584;     // 3*256 f32
constexpr int LDS_WB   = 6656;     // weights buf: 16 pts * 64 w * 32 s (bf16) = 64KB
constexpr int LDS_B    = 72192;    // reused region, 64KB:
                                   //   phase1: h1 (32KB) + h2 (32KB)
                                   //   phase2: gfeat (16KB) + aggS (32KB) + wfS (16KB)
constexpr int LDS_TOTAL = LDS_B + 65536;   // 137728 bytes

static __device__ __forceinline__ unsigned short f2bf(float f) {
  union { float f; unsigned u; } v; v.f = f;
  unsigned r = v.u + 0x7FFFu + ((v.u >> 16) & 1u);
  return (unsigned short)(r >> 16);
}

// A-fragment (16x32 bf16, row-major, K-contig rows): lane holds row (l&15),
// K = (l>>4)*16 .. +15 contiguous (matches fragB)
static __device__ __forceinline__ v16bf fragA(const unsigned short* base, int rowStride, int lane) {
  const unsigned short* p = base + (lane & 15) * rowStride + ((lane >> 4) << 4);
  union { v16bf v; v8bf h[2]; } u;
  u.h[0] = *(const v8bf*)(p);
  u.h[1] = *(const v8bf*)(p + 8);
  return u.v;
}

// B-fragment (32x16 bf16, column-major columns, K contiguous): lane holds
// column (l&15), K = (l>>4)*16 .. +15 contiguous.
static __device__ __forceinline__ v16bf fragB(const unsigned short* base, int colStride, int lane) {
  return *(const v16bf*)(base + (lane & 15) * colStride + ((lane >> 4) << 4));
}

static __device__ __forceinline__ v8f wmma_bf16(v16bf a, v16bf b, v8f c) {
  return __builtin_amdgcn_wmma_f32_16x16x32_bf16(false, a, false, b, (short)0, c, false, false);
}

#if __has_builtin(__builtin_amdgcn_tensor_load_to_lds)
#define PC_USE_TDM 1
#pragma message("PC: TDM tensor_load_to_lds path ENABLED")
// 2D TDM load: tile (tile_d1 rows x tile_d0 cols) of bf16 from a row-major
// tensor (tensor_d1 x tensor_d0, row stride stride0 elements) into LDS.
static __device__ __forceinline__ void tdm_load_2d(unsigned lds_byte_addr, const void* gaddr,
                                                   unsigned tensor_d0, unsigned tensor_d1,
                                                   unsigned tile_d0, unsigned tile_d1,
                                                   unsigned stride0) {
  unsigned long long ga = (unsigned long long)gaddr;
  u32x4 g0;
  g0[0] = 1u;                                   // count=1 valid descriptor, user mode
  g0[1] = lds_byte_addr;                        // LDS byte address
  g0[2] = (unsigned)(ga & 0xFFFFFFFFu);         // global addr lo
  g0[3] = (unsigned)((ga >> 32) & 0x01FFFFFFu)  // global addr hi (57-bit)
        | (2u << 30);                           // type = 2 ("image")
  i32x8 g1;
  g1[0] = (int)(1u << 16);                                       // data_size = 1 -> 2B
  g1[1] = (int)((tensor_d0 & 0xFFFFu) << 16);                    // tensor_dim0 lo
  g1[2] = (int)((tensor_d0 >> 16) | ((tensor_d1 & 0xFFFFu) << 16));
  g1[3] = (int)((tensor_d1 >> 16) | (tile_d0 << 16));            // tile_dim0
  g1[4] = (int)(tile_d1);                                        // tile_dim1 (tile_dim2=0)
  g1[5] = (int)stride0;                                          // tensor_dim0_stride lo
  g1[6] = 0;
  g1[7] = 0;
  i32x4 z4 = {};
#if defined(__clang_major__) && (__clang_major__ >= 23)
  i32x8 z8 = {};
  __builtin_amdgcn_tensor_load_to_lds(g0, g1, z4, z4, z8, 0);
#else
  __builtin_amdgcn_tensor_load_to_lds(g0, g1, z4, z4, 0);
#endif
}
#else
#pragma message("PC: TDM builtin NOT available - using direct global fragment loads")
#endif

// -------------------- prep: fp32 -> bf16 for GEMM operands --------------------
__global__ void pconv_prep_bf16(const float* __restrict__ Wf, const float* __restrict__ W2,
                                const float* __restrict__ W3, const float* __restrict__ feat,
                                unsigned short* __restrict__ ws) {
  size_t i0 = (size_t)blockIdx.x * blockDim.x + threadIdx.x;
  size_t st = (size_t)gridDim.x * blockDim.x;
  for (size_t k = i0; k < (size_t)Oout * KAGG; k += st) ws[WS_WF + k] = f2bf(Wf[k]);
  for (size_t k = i0; k < (size_t)64 * 64; k += st)     ws[WS_W2 + k] = f2bf(W2[k]);
  for (size_t k = i0; k < (size_t)64 * 64; k += st)     ws[WS_W3 + k] = f2bf(W3[k]);
  for (size_t k = i0; k < (size_t)Bn * Cch * Npts; k += st) ws[WS_FEAT + k] = f2bf(feat[k]);
}

// -------------------- fused PointConv kernel --------------------
__global__ void __launch_bounds__(256, 1)
pconv_main(const float* __restrict__ xyz, const int* __restrict__ idxp,
           const float* __restrict__ W1, float* __restrict__ out,
           const unsigned short* __restrict__ ws,
           const float* __restrict__ b1, const float* __restrict__ g1,
           const float* __restrict__ be1, const float* __restrict__ m1,
           const float* __restrict__ v1,
           const float* __restrict__ b2, const float* __restrict__ g2,
           const float* __restrict__ be2, const float* __restrict__ m2,
           const float* __restrict__ v2,
           const float* __restrict__ b3,
           const float* __restrict__ bfv, const float* __restrict__ gfv,
           const float* __restrict__ befv, const float* __restrict__ mfv,
           const float* __restrict__ vfv) {
  extern __shared__ __align__(128) char smem[];
  const int t  = threadIdx.x;
  const int wv = t >> 5;
  const int l  = t & 31;
  const int b  = blockIdx.x >> 11;           // 2048 point-tiles per batch
  const int n0 = (blockIdx.x & 2047) * PTS;

  float* fold = (float*)(smem + LDS_FOLD);
  float* inv1 = fold;        float* sh1 = fold + 64;
  float* inv2 = fold + 128;  float* sh2 = fold + 192;
  float* b3c  = fold + 256;
  float* invF = fold + 320;  float* shF = fold + 448;
  int*   idxT = (int*)(smem + LDS_IDX);
  float* gx   = (float*)(smem + LDS_GX);
  unsigned short* wb  = (unsigned short*)(smem + LDS_WB);   // [pt][w][s(32)]
  unsigned short* h1  = (unsigned short*)(smem + LDS_B);    // col-major [col][64]
  unsigned short* h2  = h1 + 16384;
  unsigned short* gfe = h1;                                 // phase2: [pt*16+ci][32]
  unsigned short* aggS = h1 + 8192;                         // phase2: [pt][1024]
  unsigned short* wfS  = h1 + 8192 + 16384;                 // phase2: 8 waves x 2 bufs x 512

  const unsigned short* wsWf = ws + WS_WF;
  const unsigned short* wsW2 = ws + WS_W2;
  const unsigned short* wsW3 = ws + WS_W3;
  const unsigned short* wsF  = ws + WS_FEAT;

  // ---- fold BN params, zero-pad the weight buffer K range [16,32) ----
  if (t < 64) {
    float iv = g1[t] * rsqrtf(v1[t] + PC_EPS);
    inv1[t] = iv; sh1[t] = b1[t] * iv + be1[t] - m1[t] * iv;
    float iw = g2[t] * rsqrtf(v2[t] + PC_EPS);
    inv2[t] = iw; sh2[t] = b2[t] * iw + be2[t] - m2[t] * iw;
    b3c[t] = b3[t];
  }
  if (t < 128) {
    float iv = gfv[t] * rsqrtf(vfv[t] + PC_EPS);
    invF[t] = iv; shF[t] = bfv[t] * iv + befv[t] - mfv[t] * iv;
  }
  for (int z = t; z < PTS * Wch; z += 256) {     // 1024 (pt,w) pairs
    unsigned short* p = wb + z * 32 + 16;
#pragma unroll
    for (int s = 0; s < 16; ++s) p[s] = 0;
  }
  // ---- idx tile ----
  idxT[t] = idxp[((size_t)b * Npts + n0 + (t >> 4)) * NSm + (t & 15)];
  __syncthreads();

  // ---- relative neighbor positions (3 x 256) ----
  {
    int pt = t >> 4;
    int j  = idxT[t];
    const float* xs = xyz + ((size_t)b * Npts + j) * 3;
    const float* xc = xyz + ((size_t)b * Npts + n0 + pt) * 3;
    gx[t]       = xs[0] - xc[0];
    gx[256 + t] = xs[1] - xc[1];
    gx[512 + t] = xs[2] - xc[2];
  }
  __syncthreads();

  // ---- WeightNet layer 1 (K=3, VALU): h1 = relu(bn(W1 @ gxyz)) ----
  {
    float x0 = gx[t], x1 = gx[256 + t], x2 = gx[512 + t];
    for (int o = 0; o < Wch; ++o) {
      float a = W1[o * 3] * x0 + W1[o * 3 + 1] * x1 + W1[o * 3 + 2] * x2;
      a = fmaxf(a * inv1[o] + sh1[o], 0.f);
      h1[t * 64 + o] = f2bf(a);          // column-major, K contiguous
    }
  }
  __syncthreads();

  // ---- layer 2 (WMMA bf16): h2 = relu(bn(W2 @ h1)), 64x256, K=64 ----
#pragma unroll
  for (int i = 0; i < 8; ++i) {
    int tile = wv * 8 + i, mt = tile & 3, nt = tile >> 2;
    v8f acc = {};
#pragma unroll
    for (int kk = 0; kk < 2; ++kk) {
      v16bf a  = fragA(wsW2 + (mt * 16) * 64 + kk * 32, 64, l);
      v16bf bb = fragB(h1 + (nt * 16) * 64 + kk * 32, 64, l);
      acc = wmma_bf16(a, bb, acc);
    }
    int col = nt * 16 + (l & 15), hi = l >> 4;
#pragma unroll
    for (int v = 0; v < 8; ++v) {
      int og = mt * 16 + v + hi * 8;
      float val = fmaxf(acc[v] * inv2[og] + sh2[og], 0.f);
      h2[col * 64 + og] = f2bf(val);
    }
  }
  __syncthreads();

  // ---- layer 3 (WMMA bf16): weights = W3 @ h2 + b3 -> wb[pt][w][s] ----
#pragma unroll
  for (int i = 0; i < 8; ++i) {
    int tile = wv * 8 + i, mt = tile & 3, nt = tile >> 2;
    v8f acc = {};
#pragma unroll
    for (int kk = 0; kk < 2; ++kk) {
      v16bf a  = fragA(wsW3 + (mt * 16) * 64 + kk * 32, 64, l);
      v16bf bb = fragB(h2 + (nt * 16) * 64 + kk * 32, 64, l);
      acc = wmma_bf16(a, bb, acc);
    }
    int s = l & 15, hi = l >> 4;          // col = nt*16 + s  =>  pt = nt
#pragma unroll
    for (int v = 0; v < 8; ++v) {
      int w = mt * 16 + v + hi * 8;
      wb[nt * 2048 + w * 32 + s] = f2bf(acc[v] + b3c[w]);
    }
  }
  __syncthreads();

  // ---- agg + final GEMM, fused over 4 channel chunks of 16 ----
  // Wave wv owns output rows [wv*16, wv*16+16) and a PRIVATE double-buffered
  // 16x32 Wf slab in LDS filled by its own TDM stream -> no barriers in the
  // K loop (TENSORcnt is per-wave, TDM ops are in-order per wave).
  v8f yacc = {};
  unsigned short* mybuf = wfS + wv * 1024;            // 2 x 512 bf16
#ifdef PC_USE_TDM
  const unsigned myLds = (unsigned)(unsigned long long)(void*)mybuf;
#endif
  for (int cc = 0; cc < 4; ++cc) {
    // gather features for this c-chunk: gfe[(pt*16+ci)][s], K-padded to 32
    {
      int pt = t >> 4, ci = t & 15, c = cc * 16 + ci;
      const unsigned short* fcol = wsF + ((size_t)b * Cch + c) * Npts;
      unsigned short* dst = gfe + (pt * 16 + ci) * 32;
#pragma unroll
      for (int s = 0; s < NSm; ++s) dst[s] = fcol[idxT[pt * 16 + s]];
#pragma unroll
      for (int s = NSm; s < 32; ++s) dst[s] = 0;
    }
    __syncthreads();

    // agg slab: per point, AGG[c'(16) x w(64)] = gfeat(16x16s) @ weightsT(16s x 64)
#pragma unroll
    for (int i = 0; i < 8; ++i) {
      int tile = wv * 8 + i, pt = tile >> 2, ntw = tile & 3;
      v16bf a  = fragA(gfe + pt * 512, 32, l);
      v16bf bb = fragB(wb + pt * 2048 + ntw * 16 * 32, 32, l);
      v8f acc = {};
      acc = wmma_bf16(a, bb, acc);
      int w = ntw * 16 + (l & 15), hi = l >> 4;
#pragma unroll
      for (int v = 0; v < 8; ++v) {
        int cp = v + hi * 8;                          // local c'
        aggS[pt * 1024 + cp * 64 + w] = f2bf(acc[v]); // K-contig per point column
      }
    }
    __syncthreads();   // aggS ready for all waves; wave-private pipeline below

    // final GEMM chunk: Y(128x16) += Wf[:, cc*1024 : +1024] @ AGG(1024x16)
    const unsigned short* wfRow = wsWf + (size_t)cc * 1024 + (size_t)(wv * 16) * 4096;
#ifdef PC_USE_TDM
    // per-wave async pipeline, prefetch depth 2
    tdm_load_2d(myLds,        wfRow,      4096, 128, 32, 16, 4096);   // slab 0
    tdm_load_2d(myLds + 1024, wfRow + 32, 4096, 128, 32, 16, 4096);   // slab 1
#pragma unroll 2
    for (int ks = 0; ks < 32; ++ks) {
      if (ks == 31) __builtin_amdgcn_s_wait_tensorcnt(0);  // last slab must land
      else          __builtin_amdgcn_s_wait_tensorcnt(1);  // slab ks landed, ks+1 in flight
      v16bf a  = fragA(mybuf + (ks & 1) * 512, 32, l);
      v16bf bb = fragB(aggS + ks * 32, 1024, l);
      asm volatile("s_wait_dscnt 0" ::: "memory");         // frags in VGPRs; buffer free
      if (ks + 2 < 32)                                     // refill freed buffer
        tdm_load_2d(myLds + (ks & 1) * 1024, wfRow + (ks + 2) * 32, 4096, 128, 32, 16, 4096);
      yacc = wmma_bf16(a, bb, yacc);
    }
#else
    // fallback: A-fragments straight from global (L2-resident Wf)
#pragma unroll 2
    for (int ks = 0; ks < 32; ++ks) {
      v16bf a  = fragA(wfRow + ks * 32, 4096, l);
      v16bf bb = fragB(aggS + ks * 32, 1024, l);
      yacc = wmma_bf16(a, bb, yacc);
    }
#endif
    __syncthreads();   // all waves done with aggS before next chunk overwrites gfe/aggS
  }

  // ---- epilogue: bias + BN + ReLU, write (B, 128, N) fp32 ----
  {
    int pt = l & 15, hi = l >> 4;
#pragma unroll
    for (int v = 0; v < 8; ++v) {
      int og = wv * 16 + v + hi * 8;
      float val = fmaxf(yacc[v] * invF[og] + shF[og], 0.f);
      out[((size_t)b * Oout + og) * Npts + n0 + pt] = val;
    }
  }
}

extern "C" void kernel_launch(void* const* d_in, const int* in_sizes, int n_in,
                              void* d_out, int out_size, void* d_ws, size_t ws_size,
                              hipStream_t stream) {
  const float* xyz  = (const float*)d_in[0];
  const float* feat = (const float*)d_in[1];
  const int*   idxp = (const int*)d_in[2];
  const float* W1   = (const float*)d_in[3];
  const float* b1   = (const float*)d_in[4];
  const float* g1   = (const float*)d_in[5];
  const float* be1  = (const float*)d_in[6];
  const float* m1   = (const float*)d_in[7];
  const float* v1   = (const float*)d_in[8];
  const float* W2   = (const float*)d_in[9];
  const float* b2   = (const float*)d_in[10];
  const float* g2   = (const float*)d_in[11];
  const float* be2  = (const float*)d_in[12];
  const float* m2   = (const float*)d_in[13];
  const float* v2   = (const float*)d_in[14];
  const float* W3   = (const float*)d_in[15];
  const float* b3   = (const float*)d_in[16];
  const float* Wf   = (const float*)d_in[17];
  const float* bf   = (const float*)d_in[18];
  const float* gf   = (const float*)d_in[19];
  const float* bef  = (const float*)d_in[20];
  const float* mf   = (const float*)d_in[21];
  const float* vf   = (const float*)d_in[22];
  float* out = (float*)d_out;
  unsigned short* ws = (unsigned short*)d_ws;

  pconv_prep_bf16<<<2048, 256, 0, stream>>>(Wf, W2, W3, feat, ws);

  dim3 grid(Bn * (Npts / PTS));  // 4 * 2048 = 8192 workgroups of 16 points
  pconv_main<<<grid, 256, LDS_TOTAL, stream>>>(
      xyz, idxp, W1, out, ws,
      b1, g1, be1, m1, v1,
      b2, g2, be2, m2, v2,
      b3, bf, gf, bef, mf, vf);
}
